// NeuroKernel_5884105195994
// MI455X (gfx1250) — matrix operand order repaired and checked by reading, not compiled, not run on hardware
//
#include <hip/hip_runtime.h>

typedef __attribute__((ext_vector_type(16))) _Float16 v16h;
typedef __attribute__((ext_vector_type(8)))  float    v8f;

#define NPT 512

// ---------------------------------------------------------------------------
// Kernel 1: repack W2 (1024x128 f32, row-major) into f16 "fragment-ready"
// layout: W2p[(kc*8+nt)*32 + lane][t] where, in the k order this code uses (same as A) for the B matrix of
// v_wmma_f32_16x16x32_f16 lane<16 holds column (lane), K = t (0..15);
// lane>=16 holds column (lane-16), K = 16+t.  One v16h (32B) per lane,
// perfectly coalesced at fragment-load time, L2-resident (256 KB).
// ---------------------------------------------------------------------------
__global__ __launch_bounds__(256) void pack_w2_kernel(
    const float* __restrict__ W2, _Float16* __restrict__ W2p) {
  int g  = blockIdx.x * 256 + threadIdx.x;   // 0..8191 lane-slots
  int L  = g & 31;
  int bt = g >> 5;                           // bt = kc*8 + nt, 0..255
  int kc = bt >> 3;
  int nt = bt & 7;
  int c  = L & 15;
  int hi = L >> 4;
  _Float16* dst = W2p + (size_t)g * 16;
#pragma unroll
  for (int t = 0; t < 16; ++t) {
    int k = t + (hi << 4);                   // B fragment K map
    dst[t] = (_Float16)W2[(size_t)(kc * 32 + k) * 128 + nt * 16 + c];
  }
}

// ---------------------------------------------------------------------------
// Kernel 2: fused pairwise MLP.  Block = fixed i, j in [j0, j0+128).
// Layer 1 is separable: h1[j,h] = relu(x[i]*W1[0,h] + x[j]*W1[1,h] + b1[h]).
// The 262144x1024 @ 1024x128 layer-2 GEMM is done with v_wmma_f32_16x16x32_f16:
// per wave a 16x128 output strip, K looped in 32-wide chunks, A fragment
// built in registers (layer-1 + relu + f16 cvt fused), B fragment a single
// coalesced v16h load from the packed W2.  Epilogue fuses b2+relu, the
// 128->1 W3 GEMV (shfl-xor reduction), b3 and the triu mask; writes K (f32).
// ---------------------------------------------------------------------------
__global__ __launch_bounds__(256) void mlp_pairs_kernel(
    const float* __restrict__ x,  const float* __restrict__ W1,
    const float* __restrict__ b1, const float* __restrict__ b2,
    const float* __restrict__ W3, const float* __restrict__ b3,
    const _Float16* __restrict__ W2p, float* __restrict__ Kf) {
  const int i   = blockIdx.x >> 2;
  const int j0  = (blockIdx.x & 3) << 7;
  const int tid = threadIdx.x;

  // Entire tile below the diagonal: K rows are zero, skip the GEMM.
  if (j0 + 127 < i) {
    if (tid < 128) Kf[i * NPT + j0 + tid] = 0.f;
    return;
  }

  __shared__ float cbs[128][33];   // x[j]*W1[1,h]+b1[h] for current K chunk
  __shared__ float a_s[32];        // x[i]*W1[0,h]

  const int L  = tid & 31;         // lane in wave32
  const int w  = tid >> 5;         // wave id 0..7 -> rows w*16..w*16+15
  const int r  = L & 15;
  const int hi = L >> 4;
  const float xi = x[i];

  v8f acc[8];
#pragma unroll
  for (int n = 0; n < 8; ++n)
#pragma unroll
    for (int v = 0; v < 8; ++v) acc[n][v] = 0.f;

  const v16h* __restrict__ W2pv = (const v16h*)W2p;

  for (int kc = 0; kc < 32; ++kc) {
    const int k0 = kc << 5;
    __syncthreads();
#pragma unroll
    for (int it = 0; it < 16; ++it) {
      int e  = tid + it * 256;          // 0..4095
      int rr = e >> 5, kk = e & 31;
      cbs[rr][kk] = x[j0 + rr] * W1[1024 + k0 + kk] + b1[k0 + kk];
    }
    if (tid < 32) a_s[tid] = xi * W1[k0 + tid];
    __syncthreads();

    // A fragment (16x32 f16): lane<16 -> K in {0..15}, lane>=16 -> K in {16..31}
    v16h af;
#pragma unroll
    for (int t = 0; t < 16; ++t) {
      int k   = t + (hi << 4);
      float v = a_s[k] + cbs[(w << 4) + r][k];
      af[t]   = (_Float16)(v > 0.f ? v : 0.f);
    }
#pragma unroll
    for (int nt = 0; nt < 8; ++nt) {
      v16h bf = W2pv[(size_t)((kc << 3) + nt) * 32 + L];
      acc[nt] = __builtin_amdgcn_wmma_f32_16x16x32_f16(
          false, af, false, bf, (short)0, acc[nt], false, false);
    }
  }

  // Epilogue: h2 = relu(acc + b2); out = h2 . W3 + b3; K = triu(out)
  const int c = L & 15;
  float sum[8];
#pragma unroll
  for (int v = 0; v < 8; ++v) sum[v] = 0.f;
#pragma unroll
  for (int nt = 0; nt < 8; ++nt) {
    const float bb = b2[nt * 16 + c];
    const float ww = W3[nt * 16 + c];
#pragma unroll
    for (int v = 0; v < 8; ++v) {
      float h2 = acc[nt][v] + bb;
      h2 = h2 > 0.f ? h2 : 0.f;
      sum[v] += h2 * ww;
    }
  }
#pragma unroll
  for (int v = 0; v < 8; ++v) {   // reduce the 128 columns across 16 lanes
    sum[v] += __shfl_xor(sum[v], 1, 32);
    sum[v] += __shfl_xor(sum[v], 2, 32);
    sum[v] += __shfl_xor(sum[v], 4, 32);
    sum[v] += __shfl_xor(sum[v], 8, 32);
  }
  if (c == 0) {
    const float b3v = b3[0];
#pragma unroll
    for (int v = 0; v < 8; ++v) {
      int rr  = (w << 4) + (hi << 3) + v;  // C/D layout: M = v + 8*hi
      int jj  = j0 + rr;
      float val = sum[v] + b3v;
      Kf[i * NPT + jj] = (jj >= i) ? val : 0.f;
    }
  }
}

// ---------------------------------------------------------------------------
// Kernel 3: out = K^T @ K (512x512x512) with the same WMMA micro-kernel.
// out[m,n] = sum_k K[k,m]*K[k,n] -- both operands are columns of K, so both
// LDS tiles are staged k-major straight from the f32 K with f16 conversion.
// ---------------------------------------------------------------------------
__global__ __launch_bounds__(256) void ktk_kernel(
    const float* __restrict__ Kf, float* __restrict__ out) {
  const int m0  = (blockIdx.x & 3) << 7;
  const int n0  = (blockIdx.x >> 2) << 7;
  const int tid = threadIdx.x;
  const int L = tid & 31, w = tid >> 5;
  const int r = L & 15, hi = L >> 4;
  const int c = r;

  __shared__ _Float16 Am[32][136];
  __shared__ _Float16 Bn[32][136];

  v8f acc[8];
#pragma unroll
  for (int n = 0; n < 8; ++n)
#pragma unroll
    for (int v = 0; v < 8; ++v) acc[n][v] = 0.f;

  for (int kc = 0; kc < 16; ++kc) {
    const int k0 = kc << 5;
    __syncthreads();
#pragma unroll
    for (int it = 0; it < 16; ++it) {
      int e  = tid + it * 256;          // 0..4095
      int kk = e >> 7, mm = e & 127;
      Am[kk][mm] = (_Float16)Kf[(k0 + kk) * NPT + m0 + mm];
      Bn[kk][mm] = (_Float16)Kf[(k0 + kk) * NPT + n0 + mm];
    }
    __syncthreads();

    v16h af;   // A[m,k] = K[k,m]: read Am k-major per lane row
#pragma unroll
    for (int t = 0; t < 16; ++t) {
      int k = t + (hi << 4);
      af[t] = Am[k][(w << 4) + r];
    }
#pragma unroll
    for (int nt = 0; nt < 8; ++nt) {
      v16h bf;
#pragma unroll
      for (int t = 0; t < 16; ++t) {
        int k = t + (hi << 4);
        bf[t] = Bn[k][(nt << 4) + c];
      }
      acc[nt] = __builtin_amdgcn_wmma_f32_16x16x32_f16(
          false, af, false, bf, (short)0, acc[nt], false, false);
    }
  }

#pragma unroll
  for (int nt = 0; nt < 8; ++nt)
#pragma unroll
    for (int v = 0; v < 8; ++v) {
      int row = m0 + (w << 4) + (hi << 3) + v;
      int col = n0 + (nt << 4) + c;
      out[row * NPT + col] = acc[nt][v];
    }
}

// ---------------------------------------------------------------------------
extern "C" void kernel_launch(void* const* d_in, const int* in_sizes, int n_in,
                              void* d_out, int out_size, void* d_ws, size_t ws_size,
                              hipStream_t stream) {
  const float* x  = (const float*)d_in[0];
  const float* W1 = (const float*)d_in[1];
  const float* b1 = (const float*)d_in[2];
  const float* W2 = (const float*)d_in[3];
  const float* b2 = (const float*)d_in[4];
  const float* W3 = (const float*)d_in[5];
  const float* b3 = (const float*)d_in[6];
  float* out = (float*)d_out;

  char* ws = (char*)d_ws;
  _Float16* W2p = (_Float16*)ws;                 // 1024*128*2 = 256 KB
  float*    Kf  = (float*)(ws + 262144);         // 512*512*4  = 1 MB

  pack_w2_kernel<<<32, 256, 0, stream>>>(W2, W2p);
  mlp_pairs_kernel<<<2048, 256, 0, stream>>>(x, W1, b1, b2, W3, b3, W2p, Kf);
  ktk_kernel<<<16, 256, 0, stream>>>(Kf, out);
}
